// StandardDecentralizedModel_33277406609828
// MI455X (gfx1250) — matrix operand order repaired and checked by reading, not compiled, not run on hardware
//
#include <hip/hip_runtime.h>
#include <hip/hip_bf16.h>

typedef __attribute__((ext_vector_type(16))) _Float16 v16h;
typedef __attribute__((ext_vector_type(4)))  _Float16 v4h;
typedef __attribute__((ext_vector_type(4)))  float    v4f;
typedef __attribute__((ext_vector_type(8)))  float    v8f;

// ---------------- problem constants ----------------
#define BSZ   128
#define NAG   4
#define NIMG  (BSZ*NAG)          // 512
#define HH    64
#define WW    64
#define C1    16
#define C2    32
#define HID   512
#define P1    (57*57)            // 3249 conv1 output positions
#define P2    (53*53)            // 2809 conv2 output positions
#define K1    192                // 3*8*8
#define K2    400                // 16*5*5
#define KG    96                 // 66 padded to 3*32
#define NST   (NIMG*3*HH*WW)     // 6,291,456 state elements

// ---------------- workspace layout (bytes, all 256-aligned) ----------------
#define WS_C1OUT   0                              // NIMG*C1*P1 f16 = 53,231,616
#define WS_STATESH (53231616)                     // NST f16 = 12,582,912
#define WS_FEAT    (WS_STATESH + 12582912)        // NIMG*C2 f32 = 65,536
#define WS_X       (WS_FEAT + 65536)              // NIMG*KG f16 = 98,304
#define WS_H       (WS_X + 98304)                 // NIMG*HID f16 = 524,288
#define WS_W1H     (WS_H + 524288)                // C1*K1 f16 = 6,144
#define WS_W2H     (WS_W1H + 6144)                // C2*K2 f16 = 25,600
#define WS_GWH     (WS_W2H + 25600)               // HID*KG f16 = 98,304

// A-fragment K map for 16-bit 16x32 A (ISA 7.12.2)
__device__ __forceinline__ int kmapA(int h, int hi) {
    return ((h < 8) ? h : h + 8) + (hi << 3);
}
// B-fragment K map for 16-bit 32x16 B (same K order as A)
__device__ __forceinline__ int kmapB(int h, int hi) {
    return ((h < 8) ? h : h + 8) + (hi << 3);
}

// ---- CDNA5 async global->LDS copy (ASYNCcnt-tracked, 16B per lane) ----
__device__ __forceinline__ void async_copy_b128(unsigned lds_off, const void* gsrc) {
#if defined(__gfx1250__)
    asm volatile("global_load_async_to_lds_b128 %0, %1, off"
                 :: "v"(lds_off), "v"((unsigned long long)(uintptr_t)gsrc)
                 : "memory");
#endif
}
__device__ __forceinline__ void async_wait0() {
#if defined(__gfx1250__)
    asm volatile("s_wait_asynccnt 0x0" ::: "memory");
#endif
}
__device__ __forceinline__ unsigned lds_off_u32(const void* p) {
    return (unsigned)(uintptr_t)p;   // low 32 bits of shared-aperture addr = LDS offset
}

// ---------------- prep 0: states f32 -> f16 (vectorized x4) ----------------
__global__ __launch_bounds__(256) void cvt_states_kernel(const float* __restrict__ s,
                                                         _Float16* __restrict__ sh) {
    const int t = blockIdx.x * 256 + threadIdx.x;      // NST/4 threads
    if (t >= NST / 4) return;
    v4f x = *(const v4f*)(s + (size_t)t * 4);
    v4h y = { (_Float16)x.x, (_Float16)x.y, (_Float16)x.z, (_Float16)x.w };
    *(v4h*)(sh + (size_t)t * 4) = y;
}

// ---------------- prep 1: weights f32 -> f16 (g_w zero-padded to KG) -------
__global__ __launch_bounds__(256) void cvt_weights_kernel(const float* __restrict__ w1,
                                                          const float* __restrict__ w2,
                                                          const float* __restrict__ gw,
                                                          _Float16* __restrict__ w1h,
                                                          _Float16* __restrict__ w2h,
                                                          _Float16* __restrict__ gwh) {
    const int t = blockIdx.x * 256 + threadIdx.x;
    if (t < C1 * K1) {
        w1h[t] = (_Float16)w1[t];
    } else if (t < C1 * K1 + C2 * K2) {
        const int i = t - C1 * K1;
        w2h[i] = (_Float16)w2[i];
    } else if (t < C1 * K1 + C2 * K2 + HID * KG) {
        const int i = t - C1 * K1 - C2 * K2;
        const int n = i / KG, k = i % KG;
        gwh[i] = (k < 66) ? (_Float16)gw[n * 66 + k] : (_Float16)0.0f;
    }
}

// ---------------- prep 2: zero the max-pool accumulator ----------------
__global__ __launch_bounds__(256) void zero_feat_kernel(float* __restrict__ feat) {
    int t = blockIdx.x * 256 + threadIdx.x;
    if (t < NIMG * C2) feat[t] = 0.0f;
}

// ---------------- kernel 1: conv1 implicit WMMA GEMM, 32pos x 16oc / wave --
// grid.x = NIMG * ceil(P1/256); block = 256 (8 waves x 32 positions)
__global__ __launch_bounds__(256) void conv1_kernel(const _Float16* __restrict__ sh,
                                                    const _Float16* __restrict__ w1h,
                                                    const float* __restrict__ b1,
                                                    _Float16* __restrict__ c1out) {
    __shared__ __align__(16) _Float16 ldsW[C1 * K1];   // 6 KB
    const int tid = threadIdx.x;
    {
        const unsigned base = lds_off_u32(ldsW);
        for (int i = tid; i < (C1 * K1 * 2) / 16; i += 256)
            async_copy_b128(base + i * 16, (const char*)w1h + i * 16);
        async_wait0();
    }
    __syncthreads();

    const int tilesPerImg = (P1 + 255) / 256;    // 13
    const int img  = blockIdx.x / tilesPerImg;
    const int tile = blockIdx.x % tilesPerImg;
    const int wave = tid >> 5;
    const int lane = tid & 31;
    const int nl   = lane & 15;
    const int hi   = lane >> 4;

    const int p0 = tile * 256 + wave * 32;       // two M-tiles: p0, p0+16
    int pa = p0 + nl;       if (pa > P1 - 1) pa = P1 - 1;
    int pb = p0 + 16 + nl;  if (pb > P1 - 1) pb = P1 - 1;
    const int ph0 = pa / 57, pw0 = pa % 57;
    const int ph1 = pb / 57, pw1 = pb % 57;
    const _Float16* in = sh + (size_t)img * (3 * HH * WW);

    v8f acc0 = {}, acc1 = {};
    #pragma unroll
    for (int s = 0; s < 6; ++s) {
        v16h a0, a1, b;
        #pragma unroll
        for (int h = 0; h < 16; ++h) {
            const int kA = s * 32 + kmapA(h, hi);
            const int ic = kA >> 6, rem = kA & 63, kh = rem >> 3, kw = rem & 7;
            a0[h] = in[(ic * HH + ph0 + kh) * WW + (pw0 + kw)];
            a1[h] = in[(ic * HH + ph1 + kh) * WW + (pw1 + kw)];
            b[h]  = ldsW[nl * K1 + s * 32 + kmapB(h, hi)];
        }
        acc0 = __builtin_amdgcn_wmma_f32_16x16x32_f16(false, a0, false, b,
                                                      (short)0, acc0, false, false);
        acc1 = __builtin_amdgcn_wmma_f32_16x16x32_f16(false, a1, false, b,
                                                      (short)0, acc1, false, false);
    }

    const float bo = b1[nl];
    _Float16* out = c1out + (size_t)img * (C1 * P1) + (size_t)nl * P1;
    #pragma unroll
    for (int r = 0; r < 8; ++r) {
        const int m = r + (hi << 3);
        const int q0 = p0 + m, q1 = p0 + 16 + m;
        if (q0 < P1) {
            float v = acc0[r] + bo;
            out[q0] = (_Float16)(v > 0.0f ? v : 0.0f);
        }
        if (q1 < P1) {
            float v = acc1[r] + bo;
            out[q1] = (_Float16)(v > 0.0f ? v : 0.0f);
        }
    }
}

// ---------------- kernel 2: conv2 WMMA GEMM, 16pos x 32oc / wave -----------
// grid.x = NIMG * ceil(P2/128); block = 256 (8 waves x 16 positions)
__global__ __launch_bounds__(256) void conv2_kernel(const _Float16* __restrict__ c1out,
                                                    const _Float16* __restrict__ w2h,
                                                    const float* __restrict__ b2,
                                                    float* __restrict__ feat) {
    __shared__ __align__(16) _Float16 ldsW[C2 * K2];   // 25.6 KB
    const int tid = threadIdx.x;
    {
        const unsigned base = lds_off_u32(ldsW);
        const int n128 = (C2 * K2 * 2) / 16;           // 1600
        for (int i = tid; i < n128; i += 256)
            async_copy_b128(base + i * 16, (const char*)w2h + i * 16);
        async_wait0();
    }
    __syncthreads();

    const int tilesPerImg = (P2 + 127) / 128;    // 22
    const int img  = blockIdx.x / tilesPerImg;
    const int tile = blockIdx.x % tilesPerImg;
    const int wave = tid >> 5;
    const int lane = tid & 31;
    const int nl   = lane & 15;
    const int hi   = lane >> 4;

    const int p0 = tile * 128 + wave * 16;
    int p = p0 + nl; if (p > P2 - 1) p = P2 - 1;
    const int ph = p / 53, pw = p % 53;
    const _Float16* in = c1out + (size_t)img * (C1 * P1);

    v8f acc0 = {}, acc1 = {};                    // oc 0-15, oc 16-31
    for (int s = 0; s < 13; ++s) {               // K = 400 padded to 416
        v16h a, b0, b1;
        #pragma unroll
        for (int h = 0; h < 16; ++h) {
            const int kA = s * 32 + kmapA(h, hi);
            _Float16 av = (_Float16)0.0f;
            if (kA < K2) {
                const int ic = kA / 25, rem = kA % 25, kh = rem / 5, kw = rem % 5;
                av = in[(size_t)ic * P1 + (ph + kh) * 57 + (pw + kw)];
            }
            a[h] = av;
            const int kB = s * 32 + kmapB(h, hi);
            const bool ok = (kB < K2);
            b0[h] = ok ? ldsW[nl * K2 + kB]        : (_Float16)0.0f;
            b1[h] = ok ? ldsW[(16 + nl) * K2 + kB] : (_Float16)0.0f;
        }
        acc0 = __builtin_amdgcn_wmma_f32_16x16x32_f16(false, a, false, b0,
                                                      (short)0, acc0, false, false);
        acc1 = __builtin_amdgcn_wmma_f32_16x16x32_f16(false, a, false, b1,
                                                      (short)0, acc1, false, false);
    }

    const float bo0 = b2[nl];
    const float bo1 = b2[16 + nl];
    float mx0 = 0.0f, mx1 = 0.0f;                // relu >= 0 -> 0 is identity
    #pragma unroll
    for (int r = 0; r < 8; ++r) {
        const int pp = p0 + r + (hi << 3);
        if (pp < P2) {
            float v0 = acc0[r] + bo0; v0 = v0 > 0.0f ? v0 : 0.0f;
            float v1 = acc1[r] + bo1; v1 = v1 > 0.0f ? v1 : 0.0f;
            mx0 = mx0 > v0 ? mx0 : v0;
            mx1 = mx1 > v1 ? mx1 : v1;
        }
    }
    // non-negative floats: uint bit order == float order
    atomicMax((unsigned int*)&feat[img * C2 + nl],      __float_as_uint(mx0));
    atomicMax((unsigned int*)&feat[img * C2 + 16 + nl], __float_as_uint(mx1));
}

// ---------------- kernel 3: adjacency + aggregation -> X rows (f16) --------
__global__ __launch_bounds__(256) void graph_prep_kernel(const float* __restrict__ diffs,
                                                         const float* __restrict__ feat,
                                                         _Float16* __restrict__ X) {
    const int t = blockIdx.x * 256 + threadIdx.x;
    if (t >= NIMG) return;
    const int b = t >> 2, i = t & 3;

    float lx[NAG], ly[NAG];
    #pragma unroll
    for (int j = 0; j < NAG; ++j) {
        lx[j] = diffs[(b * NAG + j) * 2 + 0];
        ly[j] = diffs[(b * NAG + j) * 2 + 1];
    }
    const float thresh = 1.41421356237f;         // ||(1,1)||
    float aij[NAG], rs = 0.0f;
    #pragma unroll
    for (int j = 0; j < NAG; ++j) {
        const float dx = lx[i] - lx[j], dy = ly[i] - ly[j];
        const float n = sqrtf(dx * dx + dy * dy);
        aij[j] = (n < thresh) ? 1.0f : 0.0f;
        rs += aij[j];
    }
    rs = fmaxf(rs, 1e-6f);
    float Am[NAG];
    #pragma unroll
    for (int j = 0; j < NAG; ++j) Am[j] = (j == i) ? 0.0f : aij[j] / rs;

    _Float16* Xr = X + (size_t)t * KG;
    for (int c = 0; c < C2; ++c) {
        Xr[c] = (_Float16)feat[t * C2 + c];
        float fa = 0.0f;
        #pragma unroll
        for (int j = 0; j < NAG; ++j) fa += Am[j] * feat[(b * NAG + j) * C2 + c];
        Xr[C2 + c] = (_Float16)fa;
    }
    float dax = 0.0f, day = 0.0f;
    #pragma unroll
    for (int j = 0; j < NAG; ++j) { dax += Am[j] * (lx[i] - lx[j]); day += Am[j] * (ly[i] - ly[j]); }
    Xr[64] = (_Float16)dax;
    Xr[65] = (_Float16)day;
    for (int c = 66; c < KG; ++c) Xr[c] = (_Float16)0.0f;
}

// ---------------- kernel 4: graph layer GEMM 512x512xK96 via WMMA ----------
// grid = 64 blocks; block 256 = 8 waves, each 16 rows x 32 hid (2 wmma/step)
__global__ __launch_bounds__(256) void gemm_g_kernel(const _Float16* __restrict__ X,
                                                     const _Float16* __restrict__ gwh,
                                                     const float* __restrict__ gb,
                                                     _Float16* __restrict__ hbuf) {
    __shared__ __align__(16) _Float16 ldsB[64 * KG];   // 12 KB
    const int tid = threadIdx.x;
    const int bn = blockIdx.x & 7;               // 8 N-blocks of 64
    const int bm = blockIdx.x >> 3;              // 8 M-blocks of 64
    {
        const unsigned base = lds_off_u32(ldsB);
        const _Float16* src = gwh + (size_t)bn * 64 * KG;   // 64 contiguous rows
        for (int i = tid; i < (64 * KG * 2) / 16; i += 256)
            async_copy_b128(base + i * 16, (const char*)src + i * 16);
        async_wait0();
    }
    __syncthreads();

    const int wave = tid >> 5;
    const int lane = tid & 31;
    const int nl   = lane & 15;
    const int hi   = lane >> 4;
    const int M0 = bm * 64 + (wave >> 1) * 16;   // 4 M-tiles per block
    const int wn = (wave & 1) * 32;              // two 16-wide B tiles per wave

    v8f acc0 = {}, acc1 = {};
    #pragma unroll
    for (int s = 0; s < 3; ++s) {
        v16h a, b0, b1;
        #pragma unroll
        for (int h = 0; h < 16; ++h) {
            a[h]  = X[(size_t)(M0 + nl) * KG + s * 32 + kmapA(h, hi)];
            const int kB = s * 32 + kmapB(h, hi);
            b0[h] = ldsB[(wn + nl) * KG + kB];
            b1[h] = ldsB[(wn + 16 + nl) * KG + kB];
        }
        acc0 = __builtin_amdgcn_wmma_f32_16x16x32_f16(false, a, false, b0,
                                                      (short)0, acc0, false, false);
        acc1 = __builtin_amdgcn_wmma_f32_16x16x32_f16(false, a, false, b1,
                                                      (short)0, acc1, false, false);
    }

    const int n0 = bn * 64 + wn + nl;
    const float bo0 = gb[n0];
    const float bo1 = gb[n0 + 16];
    #pragma unroll
    for (int r = 0; r < 8; ++r) {
        const int m = M0 + r + (hi << 3);
        float v0 = acc0[r] + bo0;
        float v1 = acc1[r] + bo1;
        hbuf[(size_t)m * HID + n0]      = (_Float16)(v0 > 0.0f ? v0 : 0.0f);
        hbuf[(size_t)m * HID + n0 + 16] = (_Float16)(v1 > 0.0f ? v1 : 0.0f);
    }
}

// ---------------- kernel 5: fc4 / fc5 heads ----------------
__global__ __launch_bounds__(256) void head_kernel(const _Float16* __restrict__ hbuf,
                                                   const float* __restrict__ fc4w,
                                                   const float* __restrict__ fc4b,
                                                   const float* __restrict__ fc5w,
                                                   const float* __restrict__ fc5b,
                                                   float* __restrict__ out) {
    const int t = blockIdx.x * 256 + threadIdx.x;
    if (t >= NIMG * 3) return;
    const int row = t / 3, o = t % 3;
    const float* wv = (o < 2) ? (fc4w + o * HID) : fc5w;
    float s = (o < 2) ? fc4b[o] : fc5b[0];
    const _Float16* hr = hbuf + (size_t)row * HID;
    for (int k = 0; k < HID; ++k) s += (float)hr[k] * wv[k];
    if (o < 2) out[row * 2 + o] = s;             // policies (128,4,2)
    else       out[NIMG * 2 + row] = s;          // values   (128,4,1)
}

extern "C" void kernel_launch(void* const* d_in, const int* in_sizes, int n_in,
                              void* d_out, int out_size, void* d_ws, size_t ws_size,
                              hipStream_t stream) {
    const float* diffs  = (const float*)d_in[0];
    const float* states = (const float*)d_in[1];
    const float* w1     = (const float*)d_in[2];
    const float* b1     = (const float*)d_in[3];
    const float* w2     = (const float*)d_in[4];
    const float* b2     = (const float*)d_in[5];
    const float* gw     = (const float*)d_in[6];
    const float* gb     = (const float*)d_in[7];
    const float* fc4w   = (const float*)d_in[8];
    const float* fc4b   = (const float*)d_in[9];
    const float* fc5w   = (const float*)d_in[10];
    const float* fc5b   = (const float*)d_in[11];
    float* out = (float*)d_out;

    char* ws = (char*)d_ws;
    _Float16* c1out   = (_Float16*)(ws + WS_C1OUT);
    _Float16* statesH = (_Float16*)(ws + WS_STATESH);
    float*    feat    = (float*)   (ws + WS_FEAT);
    _Float16* X       = (_Float16*)(ws + WS_X);
    _Float16* hbuf    = (_Float16*)(ws + WS_H);
    _Float16* w1h     = (_Float16*)(ws + WS_W1H);
    _Float16* w2h     = (_Float16*)(ws + WS_W2H);
    _Float16* gwh     = (_Float16*)(ws + WS_GWH);

    cvt_states_kernel<<<(NST / 4 + 255) / 256, 256, 0, stream>>>(states, statesH);

    const int nw = C1 * K1 + C2 * K2 + HID * KG;       // 65,024
    cvt_weights_kernel<<<(nw + 255) / 256, 256, 0, stream>>>(w1, w2, gw, w1h, w2h, gwh);

    zero_feat_kernel<<<(NIMG * C2 + 255) / 256, 256, 0, stream>>>(feat);

    const int t1 = (P1 + 255) / 256;             // 13
    conv1_kernel<<<NIMG * t1, 256, 0, stream>>>(statesH, w1h, b1, c1out);

    const int t2 = (P2 + 127) / 128;             // 22
    conv2_kernel<<<NIMG * t2, 256, 0, stream>>>(c1out, w2h, b2, feat);

    graph_prep_kernel<<<2, 256, 0, stream>>>(diffs, feat, X);

    gemm_g_kernel<<<64, 256, 0, stream>>>(X, gwh, gb, hbuf);

    head_kernel<<<(NIMG * 3 + 255) / 256, 256, 0, stream>>>(hbuf, fc4w, fc4b,
                                                            fc5w, fc5b, out);
}
